// TerminalSecurityModel_65438121721897
// MI455X (gfx1250) — matrix operand order repaired and checked by reading, not compiled, not run on hardware
//
#include <hip/hip_runtime.h>
#include <math.h>

#define IN_DIM 128
#define HID 64

typedef __attribute__((ext_vector_type(16))) __bf16 v16bf;
typedef __attribute__((ext_vector_type(8)))  __bf16 v8bf;
typedef __attribute__((ext_vector_type(8)))  float  v8f;

__device__ __forceinline__ unsigned short f2bf(float f) {
    unsigned int u = __float_as_uint(f);
    unsigned int r = u + 0x7fffu + ((u >> 16) & 1u);   // round-to-nearest-even
    return (unsigned short)(r >> 16);
}

__device__ __forceinline__ v8bf load8bf(const unsigned short* p) {
    union { uint4 u; v8bf v; } t;
    t.u = *reinterpret_cast<const uint4*>(p);
    return t.v;
}

__device__ __forceinline__ v16bf cat16(v8bf lo, v8bf hi) {
    return __builtin_shufflevector(lo, hi, 0,1,2,3,4,5,6,7,8,9,10,11,12,13,14,15);
}

// ---------------- utility kernels ----------------

__global__ void k_zero_f32(float* p, int n) {
    int i = blockIdx.x * blockDim.x + threadIdx.x;
    if (i < n) p[i] = 0.0f;
}

__global__ void k_f32_to_bf16(const float* in, unsigned short* out, int n) {
    int i = blockIdx.x * blockDim.x + threadIdx.x;
    if (i < n) out[i] = f2bf(in[i]);
}

// wT[n*K + k] = bf16(w[k*HID + n])   (w is [K x 64] row-major, wT is [64 x K])
__global__ void k_wtrans(const float* w, unsigned short* wT, int K) {
    int i = blockIdx.x * blockDim.x + threadIdx.x;
    if (i >= 64 * K) return;
    int n = i / K, k = i - n * K;
    wT[(size_t)n * K + k] = f2bf(w[(size_t)k * HID + n]);
}

// ---------------- graph structure ----------------

__global__ void k_degree(const long long* ei, float* deg, int E) {
    int e = blockIdx.x * blockDim.x + threadIdx.x;
    if (e >= E) return;
    int dst = (int)ei[(size_t)E + e];
    atomicAdd(&deg[dst], 1.0f);
}

__global__ void k_invdeg(const float* deg, float* inv, int n) {
    int i = blockIdx.x * blockDim.x + threadIdx.x;
    if (i < n) inv[i] = 1.0f / fmaxf(deg[i], 1.0f);
}

// 16 threads/edge, float4 gather from hn[src], 4 f32 atomics into agg[dst]
__global__ void k_scatter(const float* __restrict__ hn, float* __restrict__ agg,
                          const long long* __restrict__ ei, int E) {
    int t = blockIdx.x * blockDim.x + threadIdx.x;
    int e = t >> 4;
    if (e >= E) return;
    int q = t & 15;
    int c = q * 4;
    int src = (int)ei[e];
    int dst = (int)ei[(size_t)E + e];
    float4 v = *reinterpret_cast<const float4*>(hn + (size_t)src * HID + c);
    float* a = agg + (size_t)dst * HID + c;
    atomicAdd(a + 0, v.x);
    atomicAdd(a + 1, v.y);
    atomicAdd(a + 2, v.z);
    atomicAdd(a + 3, v.w);
}

// ---------------- WMMA GEMM: Hn = A@WnT^T, Hs = A@WsT^T ----------------
// One wave computes a 16x64 output strip for one weight matrix.
__global__ void k_gemm_bf16(const unsigned short* __restrict__ A, int lda, int K,
                            const unsigned short* __restrict__ WnT,
                            const unsigned short* __restrict__ WsT,
                            float* __restrict__ Hn, float* __restrict__ Hs,
                            int Mtiles) {
    int wid  = (blockIdx.x * blockDim.x + threadIdx.x) >> 5;
    int lane = threadIdx.x & 31;
    if (wid >= Mtiles * 2) return;
    int sel   = wid & 1;
    int mtile = wid >> 1;
    const unsigned short* WT = sel ? WsT : WnT;
    float* Out = sel ? Hs : Hn;

    int m   = lane & 15;
    int hsl = lane >> 4;                    // half-wave select
    const unsigned short* arow = A + (size_t)(mtile * 16 + m) * lda;

    v8f c0 = {}, c1 = {}, c2 = {}, c3 = {};
    for (int k0 = 0; k0 < K; k0 += 32) {
        // A fragment: contiguous K per half-wave (elements 0-15: K=16*hsl..16*hsl+15)
        v16bf a = cat16(load8bf(arow + k0 + 16 * hsl),
                        load8bf(arow + k0 + 16 * hsl + 8));
        // B fragments: lane n=m, rows of WT (column of W) are contiguous
        const unsigned short* b0p = WT + (size_t)(0 * 16 + m) * K + k0 + 16 * hsl;
        const unsigned short* b1p = WT + (size_t)(1 * 16 + m) * K + k0 + 16 * hsl;
        const unsigned short* b2p = WT + (size_t)(2 * 16 + m) * K + k0 + 16 * hsl;
        const unsigned short* b3p = WT + (size_t)(3 * 16 + m) * K + k0 + 16 * hsl;
        v16bf b0 = cat16(load8bf(b0p), load8bf(b0p + 8));
        v16bf b1 = cat16(load8bf(b1p), load8bf(b1p + 8));
        v16bf b2 = cat16(load8bf(b2p), load8bf(b2p + 8));
        v16bf b3 = cat16(load8bf(b3p), load8bf(b3p + 8));
        c0 = __builtin_amdgcn_wmma_f32_16x16x32_bf16(false, a, false, b0, (short)0, c0, false, false);
        c1 = __builtin_amdgcn_wmma_f32_16x16x32_bf16(false, a, false, b1, (short)0, c1, false, false);
        c2 = __builtin_amdgcn_wmma_f32_16x16x32_bf16(false, a, false, b2, (short)0, c2, false, false);
        c3 = __builtin_amdgcn_wmma_f32_16x16x32_bf16(false, a, false, b3, (short)0, c3, false, false);
    }
    // C/D layout: VGPR r, lanes 0-15 -> M=r, lanes 16-31 -> M=r+8; N=lane&15
    size_t rowbase = (size_t)(mtile * 16 + hsl * 8) * HID;
#pragma unroll
    for (int r = 0; r < 8; ++r) {
        size_t o = rowbase + (size_t)r * HID + m;
        Out[o + 0]  = c0[r];
        Out[o + 16] = c1[r];
        Out[o + 32] = c2[r];
        Out[o + 48] = c3[r];
    }
}

// ---------------- combine + ReLU + UHG normalize (one wave per node) ----------------
__global__ void k_combine(const float* __restrict__ hsbuf, float* __restrict__ agg,
                          const float* __restrict__ invdeg,
                          float* __restrict__ hout, unsigned short* __restrict__ hbf,
                          int N) {
    int node = blockIdx.x * (blockDim.x >> 5) + (threadIdx.x >> 5);
    int lane = threadIdx.x & 31;
    if (node >= N) return;
    float id = invdeg[node];
    size_t base = (size_t)node * HID;
    float v0 = hsbuf[base + lane]      + agg[base + lane]      * id;
    float v1 = hsbuf[base + lane + 32] + agg[base + lane + 32] * id;
    v0 = fmaxf(v0, 0.0f);
    v1 = fmaxf(v1, 0.0f);
    float ss = v0 * v0 + v1 * v1;
#pragma unroll
    for (int m = 16; m > 0; m >>= 1) ss += __shfl_xor(ss, m, 32);
    float scale = 1.0f / (sqrtf(ss) + 1e-6f);
    v0 *= scale; v1 *= scale;
    hout[base + lane]      = v0;
    hout[base + lane + 32] = v1;
    hbf[base + lane]       = f2bf(v0);      // bf16 copy for next layer's GEMM (stride 64)
    hbf[base + lane + 32]  = f2bf(v1);
    agg[base + lane]       = 0.0f;          // reset accumulator for next layer
    agg[base + lane + 32]  = 0.0f;
}

// ---------------- MLP head (one wave per node) ----------------
__global__ void k_mlp(const float* __restrict__ h,
                      const float* __restrict__ mw1, const float* __restrict__ mb1,
                      const float* __restrict__ mw2, const float* __restrict__ mb2,
                      float* __restrict__ out, int N) {
    int node = blockIdx.x * (blockDim.x >> 5) + (threadIdx.x >> 5);
    int lane = threadIdx.x & 31;
    if (node >= N) return;
    size_t base = (size_t)node * HID;
    float acc = mb1[lane];
#pragma unroll 8
    for (int k = 0; k < HID; ++k)
        acc = fmaf(h[base + k], mw1[(size_t)k * 32 + lane], acc);
    float p = fmaxf(acc, 0.0f) * mw2[lane];
#pragma unroll
    for (int m = 16; m > 0; m >>= 1) p += __shfl_xor(p, m, 32);
    if (lane == 0) {
        float z = p + mb2[0];
        out[node] = 1.0f / (1.0f + expf(-z));
    }
}

// ---------------- launch ----------------

extern "C" void kernel_launch(void* const* d_in, const int* in_sizes, int n_in,
                              void* d_out, int out_size, void* d_ws, size_t ws_size,
                              hipStream_t stream) {
    const float*     x   = (const float*)d_in[0];
    const long long* ei  = (const long long*)d_in[1];
    const float* w0s = (const float*)d_in[2];
    const float* w0n = (const float*)d_in[3];
    const float* w1s = (const float*)d_in[4];
    const float* w1n = (const float*)d_in[5];
    const float* w2s = (const float*)d_in[6];
    const float* w2n = (const float*)d_in[7];
    const float* mw1 = (const float*)d_in[8];
    const float* mb1 = (const float*)d_in[9];
    const float* mw2 = (const float*)d_in[10];
    const float* mb2 = (const float*)d_in[11];

    const int N = in_sizes[0] / IN_DIM;     // 50000
    const int E = in_sizes[1] / 2;          // 1600000

    // carve workspace (256B aligned blocks)
    char* ws = (char*)d_ws;
    auto carve = [&](size_t bytes) {
        char* p = ws;
        ws += (bytes + 255) & ~(size_t)255;
        return (void*)p;
    };
    float*          h_f32  = (float*)         carve((size_t)N * HID * 4);
    unsigned short* h_bf   = (unsigned short*)carve((size_t)N * IN_DIM * 2);
    float*          hn     = (float*)         carve((size_t)N * HID * 4);
    float*          hsb    = (float*)         carve((size_t)N * HID * 4);
    float*          agg    = (float*)         carve((size_t)N * HID * 4);
    float*          deg    = (float*)         carve((size_t)N * 4);
    float*          invdeg = (float*)         carve((size_t)N * 4);
    unsigned short* wnT    = (unsigned short*)carve((size_t)64 * IN_DIM * 2);
    unsigned short* wsT    = (unsigned short*)carve((size_t)64 * IN_DIM * 2);

    const int B = 256;

    // x -> bf16 (stride 128 for layer 0)
    k_f32_to_bf16<<<(N * IN_DIM + B - 1) / B, B, 0, stream>>>(x, h_bf, N * IN_DIM);

    // degree / inverse degree
    k_zero_f32<<<(N + B - 1) / B, B, 0, stream>>>(deg, N);
    k_degree<<<(E + B - 1) / B, B, 0, stream>>>(ei, deg, E);
    k_invdeg<<<(N + B - 1) / B, B, 0, stream>>>(deg, invdeg, N);
    k_zero_f32<<<(N * HID + B - 1) / B, B, 0, stream>>>(agg, N * HID);

    const float* Wn[3] = { w0n, w1n, w2n };
    const float* Ws[3] = { w0s, w1s, w2s };
    const int    Kl[3] = { IN_DIM, HID, HID };

    const int Mtiles = N / 16;              // 3125
    const int gemm_waves  = Mtiles * 2;
    const int gemm_blocks = (gemm_waves * 32 + 127) / 128;
    const int node_blocks = (N + 7) / 8;    // 8 waves/block

    for (int L = 0; L < 3; ++L) {
        int K = Kl[L];
        k_wtrans<<<(64 * K + B - 1) / B, B, 0, stream>>>(Wn[L], wnT, K);
        k_wtrans<<<(64 * K + B - 1) / B, B, 0, stream>>>(Ws[L], wsT, K);
        k_gemm_bf16<<<gemm_blocks, 128, 0, stream>>>(h_bf, (L == 0 ? IN_DIM : HID), K,
                                                     wnT, wsT, hn, hsb, Mtiles);
        k_scatter<<<((size_t)E * 16 + B - 1) / B, B, 0, stream>>>(hn, agg, ei, E);
        k_combine<<<node_blocks, B, 0, stream>>>(hsb, agg, invdeg, h_f32, h_bf, N);
    }

    k_mlp<<<node_blocks, B, 0, stream>>>(h_f32, mw1, mb1, mw2, mb2, (float*)d_out, N);
}
